// GATv2GCNModel_21165598835009
// MI455X (gfx1250) — compile-verified, matrix operand order checked by reading, not run on hardware
//
#include <hip/hip_runtime.h>
#include <math.h>

// ---------------- problem constants (match reference) ----------------
#define NN   30000          // nodes
#define EE   240000         // edges
#define ENL  (EE + NN)      // edges + self loops
#define BB   64             // graphs / LSTM time steps
#define HH   10             // GAT heads
#define CC   78             // per-head dim
#define DD   (HH * CC)      // 780
#define DDP  784            // DD padded to mult of 16 (GEMM N / ld)
#define KCP  96             // CC padded to mult of 32 (GEMM K)
#define DKP  800            // DD padded to mult of 32 (GEMM K for GCN)
#define SEQ  1000
#define EMB  128
#define LH   64             // LSTM hidden per direction
#define G4   (4 * LH)       // 256 gates
#define COMB 1688           // 2*780 + 128
#define K1P  1696           // COMB padded to mult of 32
#define CONVW 121
#define CONVO 32
#define FCPIN (CONVO * CONVW)   // 3872 (already mult of 32)

typedef __attribute__((ext_vector_type(16))) __bf16 v16bf;
typedef __attribute__((ext_vector_type(8)))  float  v8f;

// ---------------- helpers ----------------
__device__ __forceinline__ __bf16 f2bf(float x) {
    union { float f; unsigned u; } v; v.f = x;
    unsigned r = v.u + 0x7FFFu + ((v.u >> 16) & 1u);   // round-to-nearest-even
    unsigned short h = (unsigned short)(r >> 16);
    union { unsigned short u; __bf16 b; } cv; cv.u = h;
    return cv.b;
}
__device__ __forceinline__ __bf16 bf_zero() {
    union { unsigned short u; __bf16 b; } cv; cv.u = 0; return cv.b;
}
__device__ __forceinline__ void atomicMaxF(float* addr, float v) {
    if (v >= 0.0f) atomicMax((int*)addr, __float_as_int(v));
    else           atomicMin((unsigned int*)addr, __float_as_uint(v));
}
__device__ __forceinline__ float sigm(float x) { return 1.0f / (1.0f + __expf(-x)); }

// ---------------- utility / staging kernels ----------------
__global__ void k_fill(float* p, float v, int n) {
    int t = blockIdx.x * blockDim.x + threadIdx.x;
    if (t < n) p[t] = v;
}
// straight cast (no padding)
__global__ void k_cast(const float* __restrict__ in, __bf16* __restrict__ out, int n) {
    int t = blockIdx.x * blockDim.x + threadIdx.x;
    if (t < n) out[t] = f2bf(in[t]);
}
// row-padded cast: in [R x Cc] -> out [R x Cpad], zero-filled pad columns
__global__ void k_cast_pad(const float* __restrict__ in, __bf16* __restrict__ out,
                           int R, int Cc, int Cpad) {
    int t = blockIdx.x * blockDim.x + threadIdx.x;
    if (t >= R * Cpad) return;
    int r = t / Cpad, c = t % Cpad;
    out[t] = (c < Cc) ? f2bf(in[r * Cc + c]) : bf_zero();
}
// transposed + padded cast: in [R x Cc] (row-major) -> out Bt [Npad x Kpad],
// Bt[n*Kpad + k] = in[k*Cc + n]  (zero pad for n>=Cc or k>=R)
__global__ void k_castT_pad(const float* __restrict__ in, __bf16* __restrict__ out,
                            int R, int Cc, int Kpad, int Npad) {
    int t = blockIdx.x * blockDim.x + threadIdx.x;
    if (t >= Npad * Kpad) return;
    int n = t / Kpad, k = t % Kpad;
    out[t] = (n < Cc && k < R) ? f2bf(in[k * Cc + n]) : bf_zero();
}

// ---------------- bf16 WMMA GEMM, fully padded / unpredicated ----------------
// C[M x Npad] = act(A[M x Kpad] * Bt[Npad x Kpad]^T + bias)
// M mult of 16 (grid.y = M/16), Kpad mult of 32, Npad mult of 16.
// 4 waves / block, one 16x16 N-tile per wave. Per K-step: 4x global_load_b128 + 1 wmma.
__global__ void __launch_bounds__(128)
k_gemm_bf16(const __bf16* __restrict__ A, const __bf16* __restrict__ Bt,
            const float* __restrict__ bias, float* __restrict__ C,
            int Kpad, int Npad, int Nvalid, int act) {
    const int lane  = threadIdx.x & 31;
    const int ntile = blockIdx.x * 4 + (threadIdx.x >> 5);   // wave-uniform
    if (ntile * 16 >= Npad) return;                          // whole-wave exit
    const int n0 = ntile * 16;
    const int m0 = blockIdx.y * 16;
    const int lanelo = lane & 15;
    const int koff   = (lane < 16) ? 0 : 8;   // 16-bit operand lane striping (ISA 7.12.2)

    const __bf16* pa = A  + (size_t)(m0 + lanelo) * Kpad + koff;
    const __bf16* pb = Bt + (size_t)(n0 + lanelo) * Kpad + koff;

    v8f acc = {0.f, 0.f, 0.f, 0.f, 0.f, 0.f, 0.f, 0.f};
    for (int k0 = 0; k0 < Kpad; k0 += 32) {
        union { uint4 q[2]; v16bf v; } af, bfg;
        af.q[0]  = *(const uint4*)(pa);        // K = k0+koff   .. +7
        af.q[1]  = *(const uint4*)(pa + 16);   // K = k0+koff+16.. +23
        bfg.q[0] = *(const uint4*)(pb);
        bfg.q[1] = *(const uint4*)(pb + 16);
        acc = __builtin_amdgcn_wmma_f32_16x16x32_bf16(false, af.v, false, bfg.v,
                                                      (short)0, acc, false, false);
        pa += 32; pb += 32;
    }

    const int bcol  = n0 + lanelo;
    const int mbase = m0 + ((lane < 16) ? 0 : 8);
    const float badd = (bias != nullptr && bcol < Nvalid) ? bias[bcol] : 0.f;
#pragma unroll
    for (int r = 0; r < 8; ++r) {
        float v = acc[r] + badd;
        if (act == 1) v = v > 0.f ? v : 0.f;   // relu
        C[(size_t)(mbase + r) * Npad + bcol] = v;
    }
}

// ---------------- GATv2 edge kernels (XL/XR ld = DDP) ----------------
__device__ __forceinline__ void edge_ends(const int* __restrict__ ei, int e, int& src, int& dst) {
    if (e < EE) { src = ei[e]; dst = ei[EE + e]; }
    else        { src = e - EE; dst = e - EE; }            // self loop
}

__global__ void k_gat_logits(const float* __restrict__ XL, const float* __restrict__ XR,
                             const int* __restrict__ ei, const float* __restrict__ att,
                             float* __restrict__ logit, float* __restrict__ m,
                             float* __restrict__ deg) {
    int t = blockIdx.x * blockDim.x + threadIdx.x;
    if (t >= ENL * HH) return;
    int e = t / HH, h = t % HH;
    int src, dst; edge_ends(ei, e, src, dst);
    const float* xl = XL + (size_t)src * DDP + h * CC;
    const float* xr = XR + (size_t)dst * DDP + h * CC;
    const float* aw = att + h * CC;
    float acc = 0.f;
    for (int c = 0; c < CC; ++c) {
        float v = xl[c] + xr[c];
        v = (v > 0.f) ? v : 0.2f * v;                      // leaky_relu(0.2)
        acc += v * aw[c];
    }
    logit[t] = acc;
    atomicMaxF(&m[dst * HH + h], acc);
    if (h == 0) atomicAdd(&deg[dst], 1.0f);
}

__global__ void k_gat_num(const float* __restrict__ m, const int* __restrict__ ei,
                          float* __restrict__ logit, float* __restrict__ den) {
    int t = blockIdx.x * blockDim.x + threadIdx.x;
    if (t >= ENL * HH) return;
    int e = t / HH, h = t % HH;
    int src, dst; edge_ends(ei, e, src, dst);
    float ex = __expf(logit[t] - m[dst * HH + h]);
    logit[t] = ex;                                         // reuse buffer for exp value
    atomicAdd(&den[dst * HH + h], ex);
}

__global__ void k_gat_agg(const float* __restrict__ XL, const float* __restrict__ exv,
                          const float* __restrict__ den, const int* __restrict__ ei,
                          float* __restrict__ hacc) {
    int t = blockIdx.x * blockDim.x + threadIdx.x;
    if (t >= ENL * DD) return;
    int e = t / DD, d = t % DD, h = d / CC;
    int src, dst; edge_ends(ei, e, src, dst);
    float alpha = exv[e * HH + h] / den[dst * HH + h];
    atomicAdd(&hacc[(size_t)dst * DD + d], alpha * XL[(size_t)src * DDP + d]);
}

// h = elu(hacc + gat_b); write bf_h [NN x DKP] zero-padded for the GCN GEMM
__global__ void k_gat_fin(float* __restrict__ hacc, const float* __restrict__ gat_b,
                          __bf16* __restrict__ bf_h) {
    int t = blockIdx.x * blockDim.x + threadIdx.x;
    if (t >= NN * DKP) return;
    int n = t / DKP, d = t % DKP;
    if (d < DD) {
        float v = hacc[(size_t)n * DD + d] + gat_b[d];
        v = (v > 0.f) ? v : (__expf(v) - 1.0f);            // elu
        hacc[(size_t)n * DD + d] = v;
        bf_h[t] = f2bf(v);
    } else {
        bf_h[t] = bf_zero();
    }
}

__global__ void k_dinv(const float* __restrict__ deg, float* __restrict__ dinv) {
    int t = blockIdx.x * blockDim.x + threadIdx.x;
    if (t >= NN) return;
    float d = deg[t];
    dinv[t] = (d > 0.f) ? rsqrtf(d) : 0.f;
}

// ---------------- GCN kernels (HW/gacc ld = DDP) ----------------
__global__ void k_gcn_agg(const float* __restrict__ HW, const float* __restrict__ dinv,
                          const int* __restrict__ ei, float* __restrict__ gacc) {
    int t = blockIdx.x * blockDim.x + threadIdx.x;
    if (t >= ENL * DD) return;
    int e = t / DD, d = t % DD;
    int src, dst; edge_ends(ei, e, src, dst);
    float nrm = dinv[src] * dinv[dst];
    atomicAdd(&gacc[(size_t)dst * DDP + d], nrm * HW[(size_t)src * DDP + d]);
}

__global__ void k_gcn_fin_pool(const float* __restrict__ gacc, const float* __restrict__ gcn_b,
                               const int* __restrict__ batch,
                               float* __restrict__ gmax, float* __restrict__ gsum) {
    int t = blockIdx.x * blockDim.x + threadIdx.x;
    if (t >= NN * DD) return;
    int n = t / DD, d = t % DD;
    float g = gacc[(size_t)n * DDP + d] + gcn_b[d];
    g = (g > 0.f) ? g : 0.f;                               // relu
    int b = batch[n];
    atomicMaxF(&gmax[b * DD + d], g);
    atomicAdd(&gsum[b * DD + d], g);
}

__global__ void k_cnt(const int* __restrict__ batch, float* __restrict__ cnt) {
    int t = blockIdx.x * blockDim.x + threadIdx.x;
    if (t < NN) atomicAdd(&cnt[batch[t]], 1.0f);
}

__global__ void k_graphfeat(const float* __restrict__ gmax, const float* __restrict__ gsum,
                            const float* __restrict__ cnt, float* __restrict__ comb) {
    int t = blockIdx.x * blockDim.x + threadIdx.x;
    if (t >= BB * DD) return;
    int b = t / DD, d = t % DD;
    float c = cnt[b]; if (c < 1.0f) c = 1.0f;
    comb[b * COMB + d]      = gmax[t];
    comb[b * COMB + DD + d] = gsum[t] / c;
}

// ---------------- protein branch ----------------
__global__ void k_embed(const int* __restrict__ target, const float* __restrict__ emb,
                        __bf16* __restrict__ et_bf) {
    int t = blockIdx.x * blockDim.x + threadIdx.x;
    if (t >= BB * SEQ * EMB) return;
    int k = t % EMB, row = t / EMB;                         // row = t_step*SEQ + s (time major)
    et_bf[t] = f2bf(emb[target[row] * EMB + k]);
}

// one block per sequence position s (independent LSTM lanes); h,c live in LDS
__global__ void __launch_bounds__(G4)
k_lstm(const float* __restrict__ XG, const float* __restrict__ Whh,
       const float* __restrict__ bih, const float* __restrict__ bhh,
       float* __restrict__ hcat, int reverse, int outoff) {
    __shared__ float hbuf[LH], cbuf[LH], zbuf[G4];
    const int s = blockIdx.x;                               // 0..SEQ-1
    const int g = threadIdx.x;                              // 0..255
    if (g < LH) { hbuf[g] = 0.f; cbuf[g] = 0.f; }
    __syncthreads();
    for (int tt = 0; tt < BB; ++tt) {
        int t = reverse ? (BB - 1 - tt) : tt;
        float dot = 0.f;
#pragma unroll 8
        for (int k = 0; k < LH; ++k) dot += hbuf[k] * Whh[g * LH + k];
        zbuf[g] = XG[(size_t)(t * SEQ + s) * G4 + g] + dot + bih[g] + bhh[g];
        __syncthreads();
        if (g < LH) {
            float ig = sigm(zbuf[g]);                       // PyTorch gate order i,f,g,o
            float fg = sigm(zbuf[LH + g]);
            float gg = tanhf(zbuf[2 * LH + g]);
            float og = sigm(zbuf[3 * LH + g]);
            float c  = fg * cbuf[g] + ig * gg;
            float h  = og * tanhf(c);
            cbuf[g] = c; hbuf[g] = h;
            hcat[(size_t)(t * SEQ + s) * (2 * LH) + outoff + g] = h;
        }
        __syncthreads();
    }
}

// conv over channel dim: out[b][o][w] = sum_{ci<SEQ,k<8} hcat[b][ci][w+k]*W[o][ci][k]
__global__ void k_conv(const float* __restrict__ hcat, const float* __restrict__ Wc,
                       const float* __restrict__ bc, float* __restrict__ out) {
    int t = blockIdx.x * blockDim.x + threadIdx.x;
    if (t >= BB * CONVO * CONVW) return;
    int w = t % CONVW, o = (t / CONVW) % CONVO, b = t / (CONVW * CONVO);
    float acc = bc[o];
    for (int ci = 0; ci < SEQ; ++ci) {
        const float* hp = hcat + (size_t)(b * SEQ + ci) * (2 * LH) + w;
        const float* wp = Wc + (size_t)(o * SEQ + ci) * 8;
#pragma unroll
        for (int k = 0; k < 8; ++k) acc += hp[k] * wp[k];
    }
    out[t] = acc;
}

__global__ void k_concat_prot(const float* __restrict__ prot, float* __restrict__ comb) {
    int t = blockIdx.x * blockDim.x + threadIdx.x;
    if (t >= BB * 128) return;
    int b = t / 128, j = t % 128;
    comb[b * COMB + 2 * DD + j] = prot[t];
}

__global__ void k_out(const float* __restrict__ h2, const float* __restrict__ ow,
                      const float* __restrict__ ob, float* __restrict__ out) {
    int b = threadIdx.x;                                    // blockDim.x == BB
    float acc = ob[0];
    for (int k = 0; k < 512; ++k) acc += h2[b * 512 + k] * ow[k];
    out[b] = acc;
}

// ---------------- host launcher ----------------
static inline int cdiv(long long a, long long b) { return (int)((a + b - 1) / b); }

extern "C" void kernel_launch(void* const* d_in, const int* in_sizes, int n_in,
                              void* d_out, int out_size, void* d_ws, size_t ws_size,
                              hipStream_t stream) {
    (void)in_sizes; (void)n_in; (void)out_size; (void)ws_size;
    const float* x      = (const float*)d_in[0];
    const int*   ei     = (const int*)  d_in[1];
    const int*   batch  = (const int*)  d_in[2];
    const int*   target = (const int*)  d_in[3];
    const float* Wl     = (const float*)d_in[4];
    const float* Wr     = (const float*)d_in[5];
    const float* att    = (const float*)d_in[6];
    const float* gat_b  = (const float*)d_in[7];
    const float* gcn_w  = (const float*)d_in[8];
    const float* gcn_b  = (const float*)d_in[9];
    const float* emb    = (const float*)d_in[10];
    const float* Wih_f  = (const float*)d_in[11];
    const float* Whh_f  = (const float*)d_in[12];
    const float* bih_f  = (const float*)d_in[13];
    const float* bhh_f  = (const float*)d_in[14];
    const float* Wih_b  = (const float*)d_in[15];
    const float* Whh_b  = (const float*)d_in[16];
    const float* bih_b  = (const float*)d_in[17];
    const float* bhh_b  = (const float*)d_in[18];
    const float* conv_w = (const float*)d_in[19];
    const float* conv_b = (const float*)d_in[20];
    const float* fcp_w  = (const float*)d_in[21];
    const float* fcp_b  = (const float*)d_in[22];
    const float* fc1_w  = (const float*)d_in[23];
    const float* fc1_b  = (const float*)d_in[24];
    const float* fc2_w  = (const float*)d_in[25];
    const float* fc2_b  = (const float*)d_in[26];
    const float* out_w  = (const float*)d_in[27];
    const float* out_b  = (const float*)d_in[28];
    float* out = (float*)d_out;

    // -------- workspace carve (256B aligned) --------
    char*  base = (char*)d_ws;
    size_t off  = 0;
    auto carve = [&](size_t bytes) -> char* {
        char* p = base + off;
        off += (bytes + 255) & ~(size_t)255;
        return p;
    };
    __bf16* bf_x   = (__bf16*)carve((size_t)NN * KCP * 2);        // x padded [NN x 96]
    __bf16* bf_w   = (__bf16*)carve((size_t)4 * 1024 * 1024);     // reused Bt weight slab
    float*  XL     = (float*) carve((size_t)NN * DDP * 4);        // [NN x 784]
    float*  XR     = (float*) carve((size_t)NN * DDP * 4);
    float*  logit  = (float*) carve((size_t)ENL * HH * 4);
    float*  mbuf   = (float*) carve((size_t)NN * HH * 4);
    float*  den    = (float*) carve((size_t)NN * HH * 4);
    float*  deg    = (float*) carve((size_t)NN * 4);
    float*  dinv   = (float*) carve((size_t)NN * 4);
    float*  hacc   = (float*) carve((size_t)NN * DD * 4);
    __bf16* bf_h   = (__bf16*)carve((size_t)NN * DKP * 2);        // [NN x 800]
    float*  gmax   = (float*) carve((size_t)BB * DD * 4);
    float*  gsum   = (float*) carve((size_t)BB * DD * 4);
    float*  cnt    = (float*) carve((size_t)BB * 4);
    float*  comb   = (float*) carve((size_t)BB * COMB * 4);
    __bf16* combbf = (__bf16*)carve((size_t)BB * K1P * 2);        // [64 x 1696] padded
    float*  convo  = (float*) carve((size_t)BB * FCPIN * 4);
    __bf16* convbf = (__bf16*)carve((size_t)BB * FCPIN * 2);
    float*  prot   = (float*) carve((size_t)BB * 128 * 4);
    float*  h1     = (float*) carve((size_t)BB * 1024 * 4);
    __bf16* h1bf   = (__bf16*)carve((size_t)BB * 1024 * 2);
    float*  h2     = (float*) carve((size_t)BB * 512 * 4);
    // aliases (sequential stream order makes these safe):
    float*  HW     = XR;                      // GCN matmul result (XR dead after logits)
    float*  gacc   = XL;                      // GCN accum (XL dead after GAT aggregate)
    __bf16* et_bf  = (__bf16*)hacc;           // embeddings (hacc dead after bf_h cast)
    float*  XG     = XL;                      // LSTM gate inputs (graph branch done)
    float*  hcat   = XR;                      // BiLSTM output (HW dead after gcn_agg)

    const int T = 256;
    const float NEGINF = -__builtin_huge_valf();

    // ======== graph branch ========
    k_cast_pad<<<cdiv((long long)NN * KCP, T), T, 0, stream>>>(x, bf_x, NN, CC, KCP);
    k_castT_pad<<<cdiv(DDP * KCP, T), T, 0, stream>>>(Wl, bf_w, CC, DD, KCP, DDP);
    k_gemm_bf16<<<dim3(cdiv(DDP / 16, 4), NN / 16), 128, 0, stream>>>(bf_x, bf_w, nullptr, XL, KCP, DDP, DD, 0);
    k_castT_pad<<<cdiv(DDP * KCP, T), T, 0, stream>>>(Wr, bf_w, CC, DD, KCP, DDP);
    k_gemm_bf16<<<dim3(cdiv(DDP / 16, 4), NN / 16), 128, 0, stream>>>(bf_x, bf_w, nullptr, XR, KCP, DDP, DD, 0);

    k_fill<<<cdiv(NN * HH, T), T, 0, stream>>>(mbuf, NEGINF, NN * HH);
    k_fill<<<cdiv(NN * HH, T), T, 0, stream>>>(den, 0.f, NN * HH);
    k_fill<<<cdiv(NN, T), T, 0, stream>>>(deg, 0.f, NN);
    k_fill<<<cdiv(NN * DD, T), T, 0, stream>>>(hacc, 0.f, NN * DD);

    k_gat_logits<<<cdiv((long long)ENL * HH, T), T, 0, stream>>>(XL, XR, ei, att, logit, mbuf, deg);
    k_gat_num<<<cdiv((long long)ENL * HH, T), T, 0, stream>>>(mbuf, ei, logit, den);
    k_gat_agg<<<cdiv((long long)ENL * DD, T), T, 0, stream>>>(XL, logit, den, ei, hacc);
    k_gat_fin<<<cdiv((long long)NN * DKP, T), T, 0, stream>>>(hacc, gat_b, bf_h);
    k_dinv<<<cdiv(NN, T), T, 0, stream>>>(deg, dinv);

    k_castT_pad<<<cdiv(DDP * DKP, T), T, 0, stream>>>(gcn_w, bf_w, DD, DD, DKP, DDP);
    k_gemm_bf16<<<dim3(cdiv(DDP / 16, 4), NN / 16), 128, 0, stream>>>(bf_h, bf_w, nullptr, HW, DKP, DDP, DD, 0);

    k_fill<<<cdiv((long long)NN * DDP, T), T, 0, stream>>>(gacc, 0.f, NN * DDP);
    k_gcn_agg<<<cdiv((long long)ENL * DD, T), T, 0, stream>>>(HW, dinv, ei, gacc);

    k_fill<<<cdiv(BB * DD, T), T, 0, stream>>>(gmax, NEGINF, BB * DD);
    k_fill<<<cdiv(BB * DD, T), T, 0, stream>>>(gsum, 0.f, BB * DD);
    k_fill<<<1, BB, 0, stream>>>(cnt, 0.f, BB);
    k_gcn_fin_pool<<<cdiv((long long)NN * DD, T), T, 0, stream>>>(gacc, gcn_b, batch, gmax, gsum);
    k_cnt<<<cdiv(NN, T), T, 0, stream>>>(batch, cnt);
    k_graphfeat<<<cdiv(BB * DD, T), T, 0, stream>>>(gmax, gsum, cnt, comb);

    // ======== protein branch ========
    k_embed<<<cdiv(BB * SEQ * EMB, T), T, 0, stream>>>(target, emb, et_bf);

    // forward LSTM: gates = et @ Wih_f^T ; Wih is already Bt layout [256 x 128]
    k_cast<<<cdiv(G4 * EMB, T), T, 0, stream>>>(Wih_f, bf_w, G4 * EMB);
    k_gemm_bf16<<<dim3(cdiv(G4 / 16, 4), (BB * SEQ) / 16), 128, 0, stream>>>(et_bf, bf_w, nullptr, XG, EMB, G4, G4, 0);
    k_lstm<<<SEQ, G4, 0, stream>>>(XG, Whh_f, bih_f, bhh_f, hcat, 0, 0);

    // backward LSTM (reuse XG)
    k_cast<<<cdiv(G4 * EMB, T), T, 0, stream>>>(Wih_b, bf_w, G4 * EMB);
    k_gemm_bf16<<<dim3(cdiv(G4 / 16, 4), (BB * SEQ) / 16), 128, 0, stream>>>(et_bf, bf_w, nullptr, XG, EMB, G4, G4, 0);
    k_lstm<<<SEQ, G4, 0, stream>>>(XG, Whh_b, bih_b, bhh_b, hcat, 1, LH);

    k_conv<<<cdiv(BB * CONVO * CONVW, T), T, 0, stream>>>(hcat, conv_w, conv_b, convo);
    k_cast<<<cdiv(BB * FCPIN, T), T, 0, stream>>>(convo, convbf, BB * FCPIN);
    k_castT_pad<<<cdiv(128 * FCPIN, T), T, 0, stream>>>(fcp_w, bf_w, FCPIN, 128, FCPIN, 128);
    k_gemm_bf16<<<dim3(cdiv(128 / 16, 4), BB / 16), 128, 0, stream>>>(convbf, bf_w, fcp_b, prot, FCPIN, 128, 128, 0);
    k_concat_prot<<<cdiv(BB * 128, T), T, 0, stream>>>(prot, comb);

    // ======== combined head ========
    k_cast_pad<<<cdiv(BB * K1P, T), T, 0, stream>>>(comb, combbf, BB, COMB, K1P);
    k_castT_pad<<<cdiv(1024 * K1P, T), T, 0, stream>>>(fc1_w, bf_w, COMB, 1024, K1P, 1024);
    k_gemm_bf16<<<dim3(cdiv(1024 / 16, 4), BB / 16), 128, 0, stream>>>(combbf, bf_w, fc1_b, h1, K1P, 1024, 1024, 1);
    k_cast<<<cdiv(BB * 1024, T), T, 0, stream>>>(h1, h1bf, BB * 1024);
    k_castT_pad<<<cdiv(512 * 1024, T), T, 0, stream>>>(fc2_w, bf_w, 1024, 512, 1024, 512);
    k_gemm_bf16<<<dim3(cdiv(512 / 16, 4), BB / 16), 128, 0, stream>>>(h1bf, bf_w, fc2_b, h2, 1024, 512, 512, 1);
    k_out<<<1, BB, 0, stream>>>(h2, out_w, out_b, out);
}
